// CharRNN_76355928588466
// MI455X (gfx1250) — matrix operand order repaired and checked by reading, not compiled, not run on hardware
//
#include <hip/hip_runtime.h>
#include <hip/hip_bf16.h>

// ---------------------------------------------------------------------------
// CDNA5 (gfx1250) 2-layer LSTM char-RNN, bf16 WMMA with f32 accumulation.
//   B=64, T=512, EMB=128, H=1024, VOCAB=128
// Recurrent phase: 4-way split-K per 16x16 tile + LDS reduction to raise
// occupancy of the latency-bound 512-step chain from 256 to 1024 waves/step.
// ---------------------------------------------------------------------------

typedef __attribute__((ext_vector_type(16))) __bf16 v16bf;
typedef __attribute__((ext_vector_type(8)))  __bf16 v8bf;
typedef __attribute__((ext_vector_type(8)))  float  v8f;

#define LSTM_B    64
#define LSTM_T    512
#define LSTM_H    1024
#define LSTM_EMB  128
#define LSTM_VOC  128

// ---- WMMA fragment loaders (ISA 7.12.2 layouts, wave32) --------------------
// A 16x32 bf16: lane<16 -> M=lane,   K in {0..7,16..23}
//               lane>=16 -> M=lane-16, K in {8..15,24..31}
__device__ __forceinline__ v16bf load_frag_a(const __bf16* __restrict__ base,
                                             int lda, int lane) {
  int row = lane & 15;
  int kb  = (lane & 16) ? 8 : 0;
  const __bf16* p = base + (size_t)row * lda + kb;
  v8bf lo = *(const v8bf*)(p);        // K kb .. kb+7
  v8bf hi = *(const v8bf*)(p + 16);   // K kb+16 .. kb+23
  return __builtin_shufflevector(lo, hi, 0,1,2,3,4,5,6,7,8,9,10,11,12,13,14,15);
}

// B 32x16 bf16 (B supplied pre-transposed as BT[N][K], row n = output col):
// lane<16 -> N=lane, K in {0..7,16..23} ; lane>=16 -> N=lane-16, K in {8..15,24..31}
__device__ __forceinline__ v16bf load_frag_b(const __bf16* __restrict__ bt,
                                             int ldb, int lane) {
  int n  = lane & 15;
  int kb = (lane & 16) ? 8 : 0;
  const __bf16* p = bt + (size_t)n * ldb + kb;
  v8bf lo = *(const v8bf*)(p);
  v8bf hi = *(const v8bf*)(p + 16);
  return __builtin_shufflevector(lo, hi, 0,1,2,3,4,5,6,7,8,9,10,11,12,13,14,15);
}

__device__ __forceinline__ v8f wmma_bf16(v16bf a, v16bf b, v8f c) {
  return __builtin_amdgcn_wmma_f32_16x16x32_bf16(false, a, false, b,
                                                 (short)0, c, false, false);
}

__device__ __forceinline__ float sigf(float x) {
  return 1.0f / (1.0f + __expf(-x));
}

// ---- one-time prep kernels -------------------------------------------------
__global__ void embed_cast_kernel(const int* __restrict__ tokens,
                                  const float* __restrict__ emb,
                                  __bf16* __restrict__ x0, int total) {
  int i = blockIdx.x * blockDim.x + threadIdx.x;
  if (i >= total) return;
  int r = i >> 7, c = i & 127;                 // EMB = 128
  x0[i] = (__bf16)emb[((size_t)tokens[r] << 7) + c];
}

// in: fp32 [K][N] row-major  ->  out: bf16 [N][K] row-major (transposed)
__global__ void transpose_cast_kernel(const float* __restrict__ in,
                                      __bf16* __restrict__ out, int K, int N) {
  int i = blockIdx.x * blockDim.x + threadIdx.x;
  if (i >= K * N) return;
  int k = i / N, n = i - k * N;
  out[(size_t)n * K + k] = (__bf16)in[i];
}

// ---- C[M][N] = A[M][K] (bf16) x BT[N][K] (bf16) + bias, f32 out ------------
// One wave per (16-row M tile) x (64-col N group): 4 accumulators reuse A frag.
__global__ void gemm_bias_wmma(const __bf16* __restrict__ A,
                               const __bf16* __restrict__ BT,
                               const float* __restrict__ bias,
                               float* __restrict__ C,
                               int M, int N, int K) {
  int wave = (blockIdx.x * blockDim.x + threadIdx.x) >> 5;
  int lane = threadIdx.x & 31;
  int ngrp = N >> 6;
  int mi = wave / ngrp;
  int ng = wave - mi * ngrp;
  if (mi >= (M >> 4)) return;

  const __bf16* Ab = A + (size_t)(mi * 16) * K;
  v8f acc0 = {}, acc1 = {}, acc2 = {}, acc3 = {};
  for (int k = 0; k < K; k += 32) {
    v16bf a  = load_frag_a(Ab + k, K, lane);
    v16bf b0 = load_frag_b(BT + (size_t)(ng * 64 +  0) * K + k, K, lane);
    v16bf b1 = load_frag_b(BT + (size_t)(ng * 64 + 16) * K + k, K, lane);
    v16bf b2 = load_frag_b(BT + (size_t)(ng * 64 + 32) * K + k, K, lane);
    v16bf b3 = load_frag_b(BT + (size_t)(ng * 64 + 48) * K + k, K, lane);
    acc0 = wmma_bf16(a, b0, acc0);
    acc1 = wmma_bf16(a, b1, acc1);
    acc2 = wmma_bf16(a, b2, acc2);
    acc3 = wmma_bf16(a, b3, acc3);
  }

  int c0 = ng * 64 + (lane & 15);
  int rbase = mi * 16 + ((lane & 16) ? 8 : 0);
#pragma unroll
  for (int v = 0; v < 8; ++v) {
    float* crow = C + (size_t)(rbase + v) * N;
    crow[c0     ] = acc0[v] + bias[c0     ];
    crow[c0 + 16] = acc1[v] + bias[c0 + 16];
    crow[c0 + 32] = acc2[v] + bias[c0 + 32];
    crow[c0 + 48] = acc3[v] + bias[c0 + 48];
  }
}

// ---- fused LSTM timestep: z = xz[:,t,:] + h_{t-1} @ U ; gates ; h,c --------
// Block = 256 threads = 8 waves = 2 tile-groups x 4 K-splits.
// Each wave: 16 batch rows x 16 h-cols x 4 gates over a 256-wide K slice.
// Partials reduced through LDS; split-0 wave runs the gate epilogue.
// hseq is [B][T][H] bf16: output AND next-step A operand (lda = T*H).
__global__ void lstm_step_wmma(const float* __restrict__ xz,   // [B*T][4H]
                               __bf16* __restrict__ hseq,      // [B][T][H]
                               float*  __restrict__ cstate,    // [B][H]
                               const __bf16* __restrict__ UT,  // [4H][H]
                               int t, int first) {
  const int T = LSTM_T, H = LSTM_H;
  __shared__ v8f red[2][4][4][32];     // [group][ksplit][gate][lane] : 32 KB

  int w     = threadIdx.x >> 5;        // wave in block: 0..7
  int lane  = threadIdx.x & 31;
  int group = w >> 2;                  // tile group 0..1
  int ks    = w & 3;                   // K split 0..3
  int tile  = blockIdx.x * 2 + group;  // 0..255
  int mi    = tile >> 6;               // batch tile 0..3 (hj-major => same mi per block)
  int hj    = tile & 63;               // h-col tile 0..63

  v8f zi = {}, zf = {}, zg = {}, zo = {};
  if (!first) {
    const int lda = T * H;
    const __bf16* hp = hseq + (size_t)(mi * 16) * lda + (size_t)(t - 1) * H;
    const int k0 = ks * (H / 4), k1 = k0 + (H / 4);   // 256-wide slice, 8 iters
    for (int k = k0; k < k1; k += 32) {
      v16bf a = load_frag_a(hp + k, lda, lane);
      const __bf16* u0 = UT + (size_t)(0 * H + hj * 16) * H + k;
      const __bf16* u1 = UT + (size_t)(1 * H + hj * 16) * H + k;
      const __bf16* u2 = UT + (size_t)(2 * H + hj * 16) * H + k;
      const __bf16* u3 = UT + (size_t)(3 * H + hj * 16) * H + k;
      __builtin_prefetch(u0 + 32, 0, 1);   // global_prefetch_b8: next K slab of U
      v16bf bi = load_frag_b(u0, H, lane);
      v16bf bf = load_frag_b(u1, H, lane);
      v16bf bg = load_frag_b(u2, H, lane);
      v16bf bo = load_frag_b(u3, H, lane);
      zi = wmma_bf16(a, bi, zi);
      zf = wmma_bf16(a, bf, zf);
      zg = wmma_bf16(a, bg, zg);
      zo = wmma_bf16(a, bo, zo);
    }
  }

  // split-K reduction through LDS (ds_store_b128 / ds_load_b128)
  red[group][ks][0][lane] = zi;
  red[group][ks][1][lane] = zf;
  red[group][ks][2][lane] = zg;
  red[group][ks][3][lane] = zo;
  __syncthreads();
  if (ks != 0) return;
#pragma unroll
  for (int s = 1; s < 4; ++s) {
    zi += red[group][s][0][lane];
    zf += red[group][s][1][lane];
    zg += red[group][s][2][lane];
    zo += red[group][s][3][lane];
  }

  int col   = hj * 16 + (lane & 15);
  int rbase = mi * 16 + ((lane & 16) ? 8 : 0);
#pragma unroll
  for (int v = 0; v < 8; ++v) {
    int b = rbase + v;
    const float* xzr = xz + ((size_t)b * T + t) * (4 * H);
    float vi = zi[v] + xzr[0 * H + col];
    float vf = zf[v] + xzr[1 * H + col];
    float vg = zg[v] + xzr[2 * H + col];
    float vo = zo[v] + xzr[3 * H + col];
    float cold = first ? 0.0f : cstate[(size_t)b * H + col];
    float cnew = sigf(vf) * cold + sigf(vi) * tanhf(vg);
    cstate[(size_t)b * H + col] = cnew;
    float h = sigf(vo) * tanhf(cnew);
    hseq[((size_t)b * T + t) * H + col] = (__bf16)h;
  }
}

// ---------------------------------------------------------------------------
extern "C" void kernel_launch(void* const* d_in, const int* in_sizes, int n_in,
                              void* d_out, int out_size, void* d_ws, size_t ws_size,
                              hipStream_t stream) {
  (void)in_sizes; (void)n_in; (void)out_size; (void)ws_size;
  const int*   tokens = (const int*)  d_in[0];
  const float* emb    = (const float*)d_in[1];
  const float* W0     = (const float*)d_in[2];
  const float* U0     = (const float*)d_in[3];
  const float* b0     = (const float*)d_in[4];
  const float* W1     = (const float*)d_in[5];
  const float* U1     = (const float*)d_in[6];
  const float* b1     = (const float*)d_in[7];
  const float* Wd     = (const float*)d_in[8];
  const float* bd     = (const float*)d_in[9];
  float* out = (float*)d_out;

  const int B = LSTM_B, T = LSTM_T, H = LSTM_H, E = LSTM_EMB, V = LSTM_VOC;
  const int R = B * T;                       // 32768 GEMM rows

  // ---- workspace carve-out (256B aligned) ----
  char* wsp = (char*)d_ws;
  auto carve = [&](size_t bytes) -> void* {
    void* p = wsp; wsp += (bytes + 255) & ~(size_t)255; return p;
  };
  float*  xz    = (float*) carve((size_t)R * 4 * H * sizeof(float));   // reused L0/L1
  __bf16* hseq0 = (__bf16*)carve((size_t)R * H * sizeof(__bf16));
  __bf16* hseq1 = (__bf16*)carve((size_t)R * H * sizeof(__bf16));
  __bf16* x0    = (__bf16*)carve((size_t)R * E * sizeof(__bf16));
  __bf16* W0T   = (__bf16*)carve((size_t)4 * H * E * sizeof(__bf16));
  __bf16* U0T   = (__bf16*)carve((size_t)4 * H * H * sizeof(__bf16));
  __bf16* W1T   = (__bf16*)carve((size_t)4 * H * H * sizeof(__bf16));
  __bf16* U1T   = (__bf16*)carve((size_t)4 * H * H * sizeof(__bf16));
  __bf16* WdT   = (__bf16*)carve((size_t)V * H * sizeof(__bf16));
  float*  c0    = (float*) carve((size_t)B * H * sizeof(float));
  float*  c1    = (float*) carve((size_t)B * H * sizeof(float));

  // ---- prep: embed + cast, weight cast+transpose ----
  {
    int total = R * E;
    embed_cast_kernel<<<(total + 255) / 256, 256, 0, stream>>>(tokens, emb, x0, total);
  }
  {
    int n;
    n = E * 4 * H; transpose_cast_kernel<<<(n + 255) / 256, 256, 0, stream>>>(W0, W0T, E, 4 * H);
    n = H * 4 * H; transpose_cast_kernel<<<(n + 255) / 256, 256, 0, stream>>>(U0, U0T, H, 4 * H);
    n = H * 4 * H; transpose_cast_kernel<<<(n + 255) / 256, 256, 0, stream>>>(W1, W1T, H, 4 * H);
    n = H * 4 * H; transpose_cast_kernel<<<(n + 255) / 256, 256, 0, stream>>>(U1, U1T, H, 4 * H);
    n = H * V;     transpose_cast_kernel<<<(n + 255) / 256, 256, 0, stream>>>(Wd, WdT, H, V);
  }

  // ---- layer 0: xz = x0 @ W0 + b0 ; 512 fused recurrent steps ----
  {
    int waves = (R / 16) * (4 * H / 64);
    gemm_bias_wmma<<<(waves * 32 + 255) / 256, 256, 0, stream>>>(x0, W0T, b0, xz, R, 4 * H, E);
  }
  for (int t = 0; t < T; ++t)
    lstm_step_wmma<<<128, 256, 0, stream>>>(xz, hseq0, c0, U0T, t, t == 0);

  // ---- layer 1: xz = hseq0 @ W1 + b1 ; 512 fused recurrent steps ----
  {
    int waves = (R / 16) * (4 * H / 64);
    gemm_bias_wmma<<<(waves * 32 + 255) / 256, 256, 0, stream>>>(hseq0, W1T, b1, xz, R, 4 * H, H);
  }
  for (int t = 0; t < T; ++t)
    lstm_step_wmma<<<128, 256, 0, stream>>>(xz, hseq1, c1, U1T, t, t == 0);

  // ---- dense head: out = hseq1 @ Wd + bd (fp32 logits) ----
  {
    int waves = (R / 16) * (V / 64);
    gemm_bias_wmma<<<(waves * 32 + 255) / 256, 256, 0, stream>>>(hseq1, WdT, bd, out, R, V, H);
  }
}
